// NTXentLoss_40535901340189
// MI455X (gfx1250) — matrix operand order repaired and checked by reading, not compiled, not run on hardware
//
#include <hip/hip_runtime.h>
#include <hip/hip_bf16.h>

typedef _Float16 half_t;
typedef __attribute__((ext_vector_type(16))) _Float16 v16h;
typedef __attribute__((ext_vector_type(8)))  _Float16 v8h;
typedef __attribute__((ext_vector_type(8)))  float    v8f;

#define N2      4096          // 2N rows
#define NHALF   2048          // N
#define D       512           // embedding dim
#define KSTEPS  (D / 32)      // 16 WMMA K-steps of 32

// ---------------------------------------------------------------------------
// Zero the (single-element) output before atomic accumulation.
// ---------------------------------------------------------------------------
__global__ void zero_out_kernel(float* __restrict__ out, int n) {
  int i = blockIdx.x * blockDim.x + threadIdx.x;
  if (i < n) out[i] = 0.0f;
}

// ---------------------------------------------------------------------------
// Row-normalize x (clamped norm, as torch CosineSimilarity) and emit f16.
// One 256-thread block per row; each thread handles 2 of the 512 features.
// ---------------------------------------------------------------------------
__global__ __launch_bounds__(256) void rownorm_f16_kernel(const float* __restrict__ x,
                                                          half_t* __restrict__ xh) {
  __shared__ float red[256];
  const int row = blockIdx.x;
  const int t   = threadIdx.x;
  const float* xr = x + (size_t)row * D;
  float a0 = xr[t];
  float a1 = xr[t + 256];
  red[t] = a0 * a0 + a1 * a1;
  __syncthreads();
  for (int s = 128; s > 0; s >>= 1) {
    if (t < s) red[t] += red[t + s];
    __syncthreads();
  }
  const float inv = 1.0f / fmaxf(sqrtf(red[0]), 1e-8f);
  half_t* o = xh + (size_t)row * D;
  o[t]       = (half_t)(a0 * inv);
  o[t + 256] = (half_t)(a1 * inv);
}

// ---------------------------------------------------------------------------
// Fused Gram-matrix (WMMA f16 -> f32) + masked logsumexp + positive extraction.
// One block per 16-row strip; 8 waves each cover 32 of the 256 j-tiles.
// sim values are cosines*2, bounded in [-2,2], so plain sum-of-exp is safe.
// ---------------------------------------------------------------------------
__global__ __launch_bounds__(256) void ntxent_wmma_kernel(const half_t* __restrict__ xh,
                                                          float* __restrict__ out) {
  __shared__ __align__(32) half_t As[16 * D];   // 16 KB A-tile staging
  __shared__ float redSum[8 * 32 * 8];          // 8 KB per-(wave,lane,slot) partials
  __shared__ float posBuf[16];
  __shared__ float rowLoss[16];

  const int t    = threadIdx.x;
  const int lane = t & 31;
  const int wave = t >> 5;
  const int i0   = blockIdx.x * 16;

  // Stage the 16 x 512 f16 A-tile into LDS (row-major), 128-bit copies.
  {
    const uint4* src = (const uint4*)(xh + (size_t)i0 * D);
    uint4* dst = (uint4*)As;
    for (int idx = t; idx < (16 * D) / 8; idx += 256) dst[idx] = src[idx];
  }
  if (t < 16) posBuf[t] = 0.0f;
  __syncthreads();

  const int l15 = lane & 15;   // A row m / B-C column n within tile
  const int g   = lane >> 4;   // half-group select

  // Preload all 16 A fragments (contiguous k order per lane half, not the ISA table order):
  // lane m = l15; halves 0..15 -> K = kk + 16g .. +15 (contiguous, matches B)
  v16h afrag[KSTEPS];
#pragma unroll
  for (int s = 0; s < KSTEPS; ++s) {
    const half_t* ap = As + l15 * D + s * 32;
    afrag[s] = *(const v16h*)(ap + 16 * g);
  }

  float sumExp[8];
#pragma unroll
  for (int r = 0; r < 8; ++r) sumExp[r] = 0.0f;

  for (int jt = wave; jt < N2 / 16; jt += 8) {
    const int j0 = jt * 16;
    // B fragment: lane holds column (j0+l15); K halves kk+16g .. kk+16g+15 are
    // contiguous along that row of xh -> one 32-byte load per lane per K-step.
    const half_t* bbase = xh + (size_t)(j0 + l15) * D + 16 * g;
    v8f acc = {};
#pragma unroll
    for (int s = 0; s < KSTEPS; ++s) {
      v16h b = *(const v16h*)(bbase + s * 32);
      acc = __builtin_amdgcn_wmma_f32_16x16x32_f16(
          /*neg_a=*/false, afrag[s], /*neg_b=*/false, b,
          /*c_mod=*/(short)0, acc, /*reuse_a=*/false, /*reuse_b=*/false);
    }
    // Epilogue: C/D layout -> lane slot r is (row = r + 8g, col = l15).
    const int j = j0 + l15;
#pragma unroll
    for (int r = 0; r < 8; ++r) {
      const float c   = acc[r] * 2.0f;          // divide by TEMPERATURE = 0.5
      const int   row = i0 + r + 8 * g;         // global row index i
      if (j == ((row + NHALF) & (N2 - 1))) posBuf[r + 8 * g] = c;  // positive logit
      if (j != row) sumExp[r] += __expf(c);     // exclude diagonal from lse
    }
  }

  // Dump per-(wave,lane,slot) partial sums, then combine per row.
  {
    const int base = (wave * 32 + lane) * 8;
#pragma unroll
    for (int r = 0; r < 8; ++r) redSum[base + r] = sumExp[r];
  }
  __syncthreads();

  if (t < 16) {
    // Local row t lives in slot r = t&7 of lanes [16*(t>>3), 16*(t>>3)+15].
    const int g2 = t >> 3;
    const int r  = t & 7;
    float s = 0.0f;
    for (int w = 0; w < 8; ++w)
      for (int l = 0; l < 16; ++l)
        s += redSum[(w * 32 + (l + 16 * g2)) * 8 + r];
    rowLoss[t] = __logf(s) - posBuf[t];
  }
  __syncthreads();
  if (t == 0) {
    float a = 0.0f;
    for (int r = 0; r < 16; ++r) a += rowLoss[r];
    atomicAdd(out, a * (1.0f / (float)N2));
  }
}

// ---------------------------------------------------------------------------
// Launch: zero output -> normalize to f16 workspace -> fused WMMA loss.
// Workspace use: 4096*512 f16 = 4 MB.
// ---------------------------------------------------------------------------
extern "C" void kernel_launch(void* const* d_in, const int* in_sizes, int n_in,
                              void* d_out, int out_size, void* d_ws, size_t ws_size,
                              hipStream_t stream) {
  const float* x  = (const float*)d_in[0];
  float* out      = (float*)d_out;
  half_t* xh      = (half_t*)d_ws;

  zero_out_kernel<<<1, 64, 0, stream>>>(out, out_size);
  rownorm_f16_kernel<<<N2, 256, 0, stream>>>(x, xh);
  ntxent_wmma_kernel<<<N2 / 16, 256, 0, stream>>>(xh, out);
}
